// MemoryAccess_87840671137817
// MI455X (gfx1250) — matrix operand order repaired and checked by reading, not compiled, not run on hardware
//
#include <hip/hip_runtime.h>
#include <hip/hip_bf16.h>
#include <math.h>

// ---- problem dims (match reference) ----
#define B_ 32
#define T_ 16
#define I_ 512
#define N_ 512
#define W_ 64
#define R_ 4
#define C_ 512          // projection columns padded 471 -> 512
#define EPSF 1e-6f
#define THREADS 512     // dnc kernel block size (16 waves of 32)

typedef _Float16 half16 __attribute__((ext_vector_type(16)));
typedef _Float16 half8  __attribute__((ext_vector_type(8)));
typedef float    float8 __attribute__((ext_vector_type(8)));

// ---------------------------------------------------------------------------
// pack x (B*T, I) f32 -> f16
// ---------------------------------------------------------------------------
__global__ __launch_bounds__(512) void pack_x_kernel(const float* __restrict__ x,
                                                     _Float16* __restrict__ Xh) {
  int idx = blockIdx.x * 512 + threadIdx.x;   // 512*512 elements
  Xh[idx] = (_Float16)x[idx];
}

// ---------------------------------------------------------------------------
// pack weights TRANSPOSED: WhT (C=512 rows x I=512 cols, f16), WhT[c][k] = W[k][c]
// so the GEMM's B-fragment per lane is two contiguous 8-half runs.
// column layout (c): [wv 0-63][ev 64-127][wg 128][ag 129][fg 130-133][rm 134-145]
//                    [wk 146-209][ws 210][rk 211-466][rs 467-470][pad 471-511]
// Also emits bias vector bcat[512] (f32).
// ---------------------------------------------------------------------------
__global__ __launch_bounds__(512) void pack_w_kernel(
    const float* __restrict__ Wwv, const float* __restrict__ Wev,
    const float* __restrict__ Wwg, const float* __restrict__ Wag,
    const float* __restrict__ Wfg, const float* __restrict__ Wrm,
    const float* __restrict__ Wwk, const float* __restrict__ Wws,
    const float* __restrict__ Wrk, const float* __restrict__ Wrs,
    const float* __restrict__ bwv, const float* __restrict__ bev,
    const float* __restrict__ bwg, const float* __restrict__ bag,
    const float* __restrict__ bfg, const float* __restrict__ brm,
    const float* __restrict__ bwk, const float* __restrict__ bws,
    const float* __restrict__ brk, const float* __restrict__ brs,
    _Float16* __restrict__ WhT, float* __restrict__ bcat) {
  int idx = blockIdx.x * 512 + threadIdx.x;   // 512*512
  int k = idx & 511, c = idx >> 9;            // consecutive threads -> consecutive k
  float v = 0.f;
  if      (c < 64)   v = Wwv[k*64 + c];
  else if (c < 128)  v = Wev[k*64 + (c-64)];
  else if (c == 128) v = Wwg[k];
  else if (c == 129) v = Wag[k];
  else if (c < 134)  v = Wfg[k*4 + (c-130)];
  else if (c < 146)  v = Wrm[k*12 + (c-134)];
  else if (c < 210)  v = Wwk[k*64 + (c-146)];
  else if (c == 210) v = Wws[k];
  else if (c < 467)  v = Wrk[k*256 + (c-211)];
  else if (c < 471)  v = Wrs[k*4 + (c-467)];
  WhT[c * 512 + k] = (_Float16)v;             // coalesced store
  if (k == 0) {
    float bv = 0.f;
    if      (c < 64)   bv = bwv[c];
    else if (c < 128)  bv = bev[c-64];
    else if (c == 128) bv = bwg[0];
    else if (c == 129) bv = bag[0];
    else if (c < 134)  bv = bfg[c-130];
    else if (c < 146)  bv = brm[c-134];
    else if (c < 210)  bv = bwk[c-146];
    else if (c == 210) bv = bws[0];
    else if (c < 467)  bv = brk[c-211];
    else if (c < 471)  bv = brs[c-467];
    bcat[c] = bv;
  }
}

// ---------------------------------------------------------------------------
// WMMA GEMM: proj(512x512,f32) = Xh(512x512,f16) @ WhT(512x512,f16)^T
// One wave per 16x16 tile; 8 waves / block; 1024 tiles -> 128 blocks.
// All fragment loads are 16B vector loads (2 per matrix per k-step).
// k-loop fully unrolled -> 16 v_wmma per wave, loads clustered in clauses.
// ---------------------------------------------------------------------------
__global__ __launch_bounds__(256) void gemm_f16_kernel(
    const _Float16* __restrict__ A, const _Float16* __restrict__ BmT,
    float* __restrict__ Cm) {
  const int wave = threadIdx.x >> 5, lane = threadIdx.x & 31;
  const int tile = blockIdx.x * 8 + wave;          // 0..1023
  const int tm = (tile >> 5) << 4;                 // tile row * 16
  const int tn = (tile & 31) << 4;                 // tile col * 16
  const int hl = lane >> 4, lr = lane & 15;

  // per-lane base pointers
  const _Float16* arow = A + (tm + lr) * 512;      // A row M = tm+lr
  const _Float16* bcol = BmT + (tn + lr) * 512;    // B col N = tn+lr (transposed)

  float8 acc = {0.f, 0.f, 0.f, 0.f, 0.f, 0.f, 0.f, 0.f};
  #pragma unroll
  for (int kc = 0; kc < 512; kc += 32) {
    // A fragment (16-bit A 16x32): lane half hl -> runs [kc+8*hl, +8) and
    // [kc+16+8*hl, +8); elements 0-7 = run0, 8-15 = run1.
    half8 a0 = *(const half8*)(arow + kc + 8 * hl);
    half8 a1 = *(const half8*)(arow + kc + 16 + 8 * hl);
    // B fragment (16-bit B 32x16): lane half hl -> K runs [kc+8*hl, +8) and
    // [kc+16+8*hl, +8), same order as A.
    half8 b0 = *(const half8*)(bcol + kc + 8 * hl);
    half8 b1 = *(const half8*)(bcol + kc + 16 + 8 * hl);
    half16 a = __builtin_shufflevector(a0, a1, 0, 1, 2, 3, 4, 5, 6, 7,
                                       8, 9, 10, 11, 12, 13, 14, 15);
    half16 bf = __builtin_shufflevector(b0, b1, 0, 1, 2, 3, 4, 5, 6, 7,
                                        8, 9, 10, 11, 12, 13, 14, 15);
    acc = __builtin_amdgcn_wmma_f32_16x16x32_f16(
        false, a, false, bf, (short)0, acc, false, false);
  }
  // D layout: VGPR v -> row = v + 8*(lane/16), col = lane%16
  #pragma unroll
  for (int v = 0; v < 8; ++v)
    Cm[(tm + v + 8 * hl) * 512 + (tn + lr)] = acc[v];
}

// ---------------------------------------------------------------------------
// block reductions (fixed 512 threads = 16 waves)
// ---------------------------------------------------------------------------
__device__ __forceinline__ float blockReduceSum(float v, volatile float* red) {
  __syncthreads();
  #pragma unroll
  for (int o = 16; o > 0; o >>= 1) v += __shfl_xor(v, o, 32);
  int w = threadIdx.x >> 5, ln = threadIdx.x & 31;
  if (ln == 0) red[w] = v;
  __syncthreads();
  if (w == 0) {
    float s = (ln < 16) ? red[ln] : 0.f;
    #pragma unroll
    for (int o = 8; o > 0; o >>= 1) s += __shfl_xor(s, o, 32);
    if (ln == 0) red[0] = s;
  }
  __syncthreads();
  return red[0];
}

__device__ __forceinline__ float blockReduceMax(float v, volatile float* red) {
  __syncthreads();
  #pragma unroll
  for (int o = 16; o > 0; o >>= 1) v = fmaxf(v, __shfl_xor(v, o, 32));
  int w = threadIdx.x >> 5, ln = threadIdx.x & 31;
  if (ln == 0) red[w] = v;
  __syncthreads();
  if (w == 0) {
    float s = (ln < 16) ? red[ln] : -3.4e38f;
    #pragma unroll
    for (int o = 8; o > 0; o >>= 1) s = fmaxf(s, __shfl_xor(s, o, 32));
    if (ln == 0) red[0] = s;
  }
  __syncthreads();
  return red[0];
}

__device__ __forceinline__ float sigmf(float x) { return 1.f / (1.f + expf(-x)); }
__device__ __forceinline__ float softplusf_(float x) {
  return (x > 20.f) ? x : log1pf(expf(x));
}

// ---------------------------------------------------------------------------
// persistent per-batch DNC scan: grid = B blocks, 512 threads each.
// memory (NxW) and link (NxN, ping-pong) live in global ws (L2-resident);
// all other recurrent state lives in LDS.
// ---------------------------------------------------------------------------
__global__ __launch_bounds__(THREADS) void dnc_kernel(
    const float* __restrict__ proj, const float* __restrict__ bcat,
    float* __restrict__ memM, float* __restrict__ link0,
    float* __restrict__ link1, float* __restrict__ out) {
  const int b = blockIdx.x;
  const int tid = threadIdx.x;
  const int lane = tid & 31, wave = tid >> 5;

  __shared__ float wvS[64], evS[64], wkS[64], rkS[4][64];
  __shared__ float wcwS[512];
  __shared__ float rcwS[4][512];
  __shared__ float allocS[512], skeyS[512];
  __shared__ int   sidxS[512];
  __shared__ float usageS[512], precS[512], wwS[512];
  __shared__ float rwpS[4][512], rwnS[4][512], fwdS[4][512], bwdS[4][512];
  __shared__ float redS[16];
  __shared__ float rmrawS[12];
  __shared__ float wgS, agS, fgS[4], wsStrS, rsS[4];
  __shared__ float bmS[4], cmS[4], fmS[4], knS[5];

  float* mem = memM + (size_t)b * N_ * W_;
  float* lA  = link0 + (size_t)b * N_ * N_;
  float* lB  = link1 + (size_t)b * N_ * N_;

  // ---- init recurrent state to zero ----
  usageS[tid] = 0.f; precS[tid] = 0.f;
  for (int i = tid; i < R_ * N_; i += THREADS) ((float*)rwpS)[i] = 0.f;
  for (int i = tid; i < N_ * W_; i += THREADS) mem[i] = 0.f;
  for (int i = tid; i < N_ * N_; i += THREADS) lA[i] = 0.f;
  __threadfence();
  __syncthreads();

  for (int t = 0; t < T_; ++t) {
    // ---- 1. projections -> activations ----
    const float* pr = proj + (size_t)(b * T_ + t) * C_;
    if (tid < 471) {
      float val = pr[tid] + bcat[tid];
      int c = tid;
      if      (c < 64)   wvS[c] = val;
      else if (c < 128)  evS[c - 64] = sigmf(val);
      else if (c == 128) wgS = sigmf(val);
      else if (c == 129) agS = sigmf(val);
      else if (c < 134)  fgS[c - 130] = sigmf(val);
      else if (c < 146)  rmrawS[c - 134] = val;
      else if (c < 210)  wkS[c - 146] = val;
      else if (c == 210) wsStrS = softplusf_(val);
      else if (c < 467)  rkS[(c - 211) >> 6][(c - 211) & 63] = val;
      else               rsS[c - 467] = softplusf_(val);
    }
    __syncthreads();
    if (tid < 4) {  // read-mode softmax over MODES=3
      float a0 = rmrawS[tid * 3], a1 = rmrawS[tid * 3 + 1], a2 = rmrawS[tid * 3 + 2];
      float m = fmaxf(a0, fmaxf(a1, a2));
      float e0 = expf(a0 - m), e1 = expf(a1 - m), e2 = expf(a2 - m);
      float s = e0 + e1 + e2;
      bmS[tid] = e0 / s; cmS[tid] = e1 / s; fmS[tid] = e2 / s;
    }
    if (tid >= 8 && tid < 13) {  // key norms
      int k = tid - 8;
      const float* kp = (k == 0) ? wkS : &rkS[k - 1][0];
      float s = 0.f;
      for (int w = 0; w < W_; ++w) s += kp[w] * kp[w];
      knS[k] = sqrtf(s);
    }
    __syncthreads();

    // ---- 2. content logits (tid == slot n, THREADS == N) ----
    {
      const int n = tid;
      float mn2 = 0.f, dw = 0.f, d0 = 0.f, d1 = 0.f, d2 = 0.f, d3 = 0.f;
      const float* mr = mem + n * W_;
      for (int w = 0; w < W_; ++w) {
        float m = mr[w];
        mn2 += m * m;
        dw += wkS[w] * m;
        d0 += rkS[0][w] * m; d1 += rkS[1][w] * m;
        d2 += rkS[2][w] * m; d3 += rkS[3][w] * m;
      }
      float mn = sqrtf(mn2);
      wcwS[n]    = wsStrS * (dw / (knS[0] * mn + EPSF));
      rcwS[0][n] = rsS[0] * (d0 / (knS[1] * mn + EPSF));
      rcwS[1][n] = rsS[1] * (d1 / (knS[2] * mn + EPSF));
      rcwS[2][n] = rsS[2] * (d2 / (knS[3] * mn + EPSF));
      rcwS[3][n] = rsS[3] * (d3 / (knS[4] * mn + EPSF));
    }
    // softmax over n for the 5 weightings
    for (int k = 0; k < 5; ++k) {
      float* L = (k == 0) ? wcwS : rcwS[k - 1];
      float x = L[tid];
      float mx = blockReduceMax(x, redS);
      float e = expf(x - mx);
      float s = blockReduceSum(e, redS);
      L[tid] = e / s;
    }

    // ---- 3. allocation: bitonic sort of nonusage (desc, idx tiebreak) ----
    __syncthreads();
    skeyS[tid] = 1.f - usageS[tid];
    sidxS[tid] = tid;
    for (int ksz = 2; ksz <= 512; ksz <<= 1)
      for (int j = ksz >> 1; j > 0; j >>= 1) {
        __syncthreads();
        int i = tid, l = i ^ j;
        if (l > i) {
          float ki = skeyS[i], kl = skeyS[l];
          int ii = sidxS[i], il = sidxS[l];
          bool iFirst = (ki > kl) || (ki == kl && ii < il);
          if (((i & ksz) == 0) ? !iFirst : iFirst) {
            skeyS[i] = kl; skeyS[l] = ki; sidxS[i] = il; sidxS[l] = ii;
          }
        }
      }
    __syncthreads();
    if (tid == 0) {  // exclusive cumprod of sorted usage, scatter back
      float p = 1.f;
      for (int i = 0; i < N_; ++i) {
        allocS[sidxS[i]] = skeyS[i] * p;
        p *= (1.f - skeyS[i]);
      }
    }
    __syncthreads();

    // ---- 4. write weighting ----
    float wwme = wgS * (agS * allocS[tid] + (1.f - agS) * wcwS[tid]);
    wwS[tid] = wwme;
    float sww = blockReduceSum(wwme, redS);

    // ---- 5. memory update (in place; old memory already consumed) ----
    for (int i = tid; i < N_ * W_; i += THREADS) {
      int n = i >> 6, w = i & 63;
      float m = mem[i];
      mem[i] = m * (1.f - wwS[n] * evS[w]) + wwS[n] * wvS[w];
    }

    // ---- 6. fused link update + fwdw/bwdw (ping-pong lA -> lB) ----
    for (int i = tid; i < R_ * N_; i += THREADS) ((float*)bwdS)[i] = 0.f;
    __threadfence();
    __syncthreads();
    {
      float bp0[16], bp1[16], bp2[16], bp3[16];
      #pragma unroll
      for (int j = 0; j < 16; ++j) { bp0[j]=0.f; bp1[j]=0.f; bp2[j]=0.f; bp3[j]=0.f; }
      for (int n = wave; n < N_; n += 16) {
        float wwn = wwS[n];
        float rn0 = rwpS[0][n], rn1 = rwpS[1][n], rn2 = rwpS[2][n], rn3 = rwpS[3][n];
        float f0 = 0.f, f1 = 0.f, f2 = 0.f, f3 = 0.f;
        if (n + 16 < N_)
          __builtin_prefetch(lA + (n + 16) * N_ + lane * 16, 0, 0);
        #pragma unroll
        for (int j = 0; j < 16; ++j) {
          int m = j * 32 + lane;
          float lold = lA[n * N_ + m];
          float lnew = (m == n) ? 0.f
                                : ((1.f - wwn - wwS[m]) * lold + wwn * precS[m]);
          lB[n * N_ + m] = lnew;
          f0 += lnew * rwpS[0][m]; f1 += lnew * rwpS[1][m];
          f2 += lnew * rwpS[2][m]; f3 += lnew * rwpS[3][m];
          bp0[j] += lnew * rn0; bp1[j] += lnew * rn1;
          bp2[j] += lnew * rn2; bp3[j] += lnew * rn3;
        }
        #pragma unroll
        for (int o = 16; o > 0; o >>= 1) {
          f0 += __shfl_xor(f0, o, 32); f1 += __shfl_xor(f1, o, 32);
          f2 += __shfl_xor(f2, o, 32); f3 += __shfl_xor(f3, o, 32);
        }
        if (lane == 0) { fwdS[0][n]=f0; fwdS[1][n]=f1; fwdS[2][n]=f2; fwdS[3][n]=f3; }
      }
      #pragma unroll
      for (int j = 0; j < 16; ++j) {
        int m = j * 32 + lane;
        atomicAdd(&bwdS[0][m], bp0[j]); atomicAdd(&bwdS[1][m], bp1[j]);
        atomicAdd(&bwdS[2][m], bp2[j]); atomicAdd(&bwdS[3][m], bp3[j]);
      }
    }
    __threadfence();
    __syncthreads();

    // ---- 7. prec + usage updates (use OLD rw_prev), 8. new read weights ----
    {
      const int n = tid;
      precS[n] = (1.f - sww) * precS[n] + wwS[n];
      float u = usageS[n];
      float uw = u + (1.f - u) * wwS[n];
      float psi = (1.f - fgS[0] * rwpS[0][n]) * (1.f - fgS[1] * rwpS[1][n]) *
                  (1.f - fgS[2] * rwpS[2][n]) * (1.f - fgS[3] * rwpS[3][n]);
      usageS[n] = uw * psi;
      #pragma unroll
      for (int r = 0; r < R_; ++r)
        rwnS[r][n] = bmS[r] * bwdS[r][n] + cmS[r] * rcwS[r][n] + fmS[r] * fwdS[r][n];
    }
    __syncthreads();

    // ---- 9. read words -> output ----
    if (tid < R_ * W_) {
      int r = tid >> 6, w = tid & 63;
      float s = 0.f;
      for (int n = 0; n < N_; ++n) s += rwnS[r][n] * mem[n * W_ + w];
      out[(((size_t)b * T_ + t) * R_ + r) * W_ + w] = s;
    }
    __syncthreads();
    for (int i = tid; i < R_ * N_; i += THREADS) ((float*)rwpS)[i] = ((float*)rwnS)[i];
    __threadfence();
    __syncthreads();
    { float* tmp = lA; lA = lB; lB = tmp; }  // ping-pong
  }
}

// ---------------------------------------------------------------------------
extern "C" void kernel_launch(void* const* d_in, const int* in_sizes, int n_in,
                              void* d_out, int out_size, void* d_ws, size_t ws_size,
                              hipStream_t stream) {
  (void)in_sizes; (void)n_in; (void)out_size; (void)ws_size;
  const float* x   = (const float*)d_in[0];
  const float* Wwv = (const float*)d_in[1];  const float* bwv = (const float*)d_in[2];
  const float* Wev = (const float*)d_in[3];  const float* bev = (const float*)d_in[4];
  const float* Wwg = (const float*)d_in[5];  const float* bwg = (const float*)d_in[6];
  const float* Wag = (const float*)d_in[7];  const float* bag = (const float*)d_in[8];
  const float* Wfg = (const float*)d_in[9];  const float* bfg = (const float*)d_in[10];
  const float* Wrm = (const float*)d_in[11]; const float* brm = (const float*)d_in[12];
  const float* Wwk = (const float*)d_in[13]; const float* bwk = (const float*)d_in[14];
  const float* Wws = (const float*)d_in[15]; const float* bws = (const float*)d_in[16];
  const float* Wrk = (const float*)d_in[17]; const float* brk = (const float*)d_in[18];
  const float* Wrs = (const float*)d_in[19]; const float* brs = (const float*)d_in[20];

  // workspace partition (all offsets 256B aligned)
  char* ws = (char*)d_ws;
  size_t off = 0;
  auto take = [&](size_t bytes) {
    char* p = ws + off;
    off = (off + bytes + 255) & ~(size_t)255;
    return p;
  };
  _Float16* Xh  = (_Float16*)take((size_t)512 * 512 * sizeof(_Float16));
  _Float16* WhT = (_Float16*)take((size_t)512 * 512 * sizeof(_Float16));
  float* bcat   = (float*)take((size_t)C_ * sizeof(float));
  float* proj   = (float*)take((size_t)512 * C_ * sizeof(float));
  float* memM   = (float*)take((size_t)B_ * N_ * W_ * sizeof(float));
  float* link0  = (float*)take((size_t)B_ * N_ * N_ * sizeof(float));
  float* link1  = (float*)take((size_t)B_ * N_ * N_ * sizeof(float));

  pack_x_kernel<<<512, 512, 0, stream>>>(x, Xh);
  pack_w_kernel<<<512, 512, 0, stream>>>(Wwv, Wev, Wwg, Wag, Wfg, Wrm, Wwk, Wws,
                                         Wrk, Wrs, bwv, bev, bwg, bag, bfg, brm,
                                         bwk, bws, brk, brs, WhT, bcat);
  gemm_f16_kernel<<<128, 256, 0, stream>>>(Xh, WhT, proj);
  dnc_kernel<<<B_, THREADS, 0, stream>>>(proj, bcat, memM, link0, link1,
                                         (float*)d_out);
}
